// SelfAttention_20761871908977
// MI455X (gfx1250) — matrix operand order repaired and checked by reading, not compiled, not run on hardware
//
#include <hip/hip_runtime.h>
#include <hip/hip_bf16.h>

// ---------------------------------------------------------------------------
// Self-attention forward, CDNA5 (gfx1250) WMMA bf16 pipeline.
//   1) cvt f32->bf16 for x, Wq, Wk, Wv, W0
//   2) Q = x@Wq^T, K = x@Wk^T  (bf16 out),  Vt = (x@Wv^T) stored [b][h][dk][s]
//   3) column-softmax stats over the QUERY axis: m[k], 1/l[k] per (b,h,k)
//   4) heads = softmax_col(QK^T/8) @ V  via score recompute + LDS repack
//   5) out = heads@W0^T (f32 out)  -- heads flat [b][h][q][dk] == reshape(b,s,d)
// Matrix op: v_wmma_f32_16x16x32_bf16 throughout.
// ---------------------------------------------------------------------------

typedef __bf16 bf16;
typedef __attribute__((ext_vector_type(16))) __bf16 v16bf;
typedef __attribute__((ext_vector_type(8)))  float  v8f;

struct alignas(16) U128 { unsigned u[4]; };
struct alignas(32) U256 { U128 lo, hi; };

static constexpr int Bm  = 4;
static constexpr int Sm  = 2048;
static constexpr int Dm  = 768;
static constexpr int NHm = 12;
static constexpr int DKm = 64;
static constexpr int Mtot = Bm * Sm;   // 8192 rows

static __device__ __forceinline__ bf16 f2bf(float f) {
  unsigned u = __builtin_bit_cast(unsigned, f);
  unsigned r = (u + 0x7FFFu + ((u >> 16) & 1u)) >> 16;   // RNE truncate
  unsigned short s = (unsigned short)r;
  return __builtin_bit_cast(bf16, s);
}

static __device__ __forceinline__ v8f wmma_bf16(v16bf a, v16bf b, v8f c) {
  // (neg_a, A, neg_b, B, c_mod, C, reuse_a, reuse_b)
  return __builtin_amdgcn_wmma_f32_16x16x32_bf16(false, a, false, b, (short)0, c,
                                                 false, false);
}

static __device__ __forceinline__ v16bf load_afrag(const bf16* row, int kk, int akb) {
  // A 16x32 bf16 fragment: lane m = lane&15; VGPR0..7 -> K 2*akb..2*akb+15
  // (akb = 0 for lanes 0-15, 8 for lanes 16-31).
  U256 a;
  a.lo = *(const U128*)(row + kk + 2 * akb);
  a.hi = *(const U128*)(row + kk + 2 * akb + 8);
  return __builtin_bit_cast(v16bf, a);
}

// ---------------------------------------------------------------------------
// 1) elementwise f32 -> bf16
// ---------------------------------------------------------------------------
__global__ __launch_bounds__(256) void cvt_f32_bf16(const float* __restrict__ in,
                                                    bf16* __restrict__ out, int n) {
  int i = blockIdx.x * blockDim.x + threadIdx.x;
  if (i < n) out[i] = f2bf(in[i]);
}

// ---------------------------------------------------------------------------
// 2/5) Y = A(bf16, M x 768) @ W(bf16, 768 x 768)^T
//   MODE 0: f32 output row-major         (final projection -> d_out)
//   MODE 1: bf16 output row-major        (Q, K)
//   MODE 2: bf16 output per-head transposed: Vt[b][h][dk][s]
// Workgroup 256 = 8 waves; wave -> 16 rows x 64 cols; K loop step 32.
// ---------------------------------------------------------------------------
template <int MODE>
__global__ __launch_bounds__(256) void gemm_xwt(const bf16* __restrict__ A,
                                                const bf16* __restrict__ W,
                                                float* __restrict__ Yf,
                                                bf16* __restrict__ Yb) {
  const int lane = threadIdx.x & 31;
  const int wave = threadIdx.x >> 5;
  const int m0 = blockIdx.x * 128 + wave * 16;
  const int n0 = blockIdx.y * 64;
  const int lm  = lane & 15;
  const int akb = (lane < 16) ? 0 : 8;    // A K-base
  const int bkb = (lane < 16) ? 0 : 16;   // B K-base
  const int rowoff = (lane < 16) ? 0 : 8; // C/D row offset

  v8f acc[4] = {};
  const bf16* arow = A + (size_t)(m0 + lm) * Dm;

  for (int kk = 0; kk < Dm; kk += 32) {
    v16bf af = load_afrag(arow, kk, akb);
#pragma unroll
    for (int t = 0; t < 4; ++t) {
      // B 32x16 fragment: lane n = lane&15, VGPR v -> K pair bkb+2v{,+1}.
      // B(k,n) = W[n][k] -> contiguous 32B along W row n.
      const bf16* wrow = W + (size_t)(n0 + t * 16 + lm) * Dm;
      v16bf bfrag = *(const v16bf*)(wrow + kk + bkb);
      acc[t] = wmma_bf16(af, bfrag, acc[t]);
    }
  }

#pragma unroll
  for (int t = 0; t < 4; ++t) {
#pragma unroll
    for (int r = 0; r < 8; ++r) {
      const int m = m0 + rowoff + r;
      const int n = n0 + t * 16 + lm;
      const float v = acc[t][r];
      if (MODE == 0) {
        Yf[(size_t)m * Dm + n] = v;
      } else if (MODE == 1) {
        Yb[(size_t)m * Dm + n] = f2bf(v);
      } else {
        const int b = m >> 11, s = m & (Sm - 1);
        const int h = n >> 6,  dk = n & 63;
        Yb[(((size_t)(b * NHm + h)) * DKm + dk) * Sm + s] = f2bf(v);
      }
    }
  }
}

// ---------------------------------------------------------------------------
// 3) Column-softmax stats: softmax is over axis=2 (the QUERY axis), i.e. each
// key column k of scores(q,k) is normalized over q. Online (flash-style)
// max/sum streaming over q-tiles; lanes (l, l+16) hold the two row-halves of a
// column and are merged with a shfl_xor(16) at the end.
// Grid: (S/128, NH, B); wave -> 16 key columns.
// ---------------------------------------------------------------------------
__global__ __launch_bounds__(256) void col_softmax_stats(const bf16* __restrict__ qb,
                                                         const bf16* __restrict__ kb,
                                                         float* __restrict__ mcol,
                                                         float* __restrict__ invl) {
  const int lane = threadIdx.x & 31;
  const int wave = threadIdx.x >> 5;
  const int b = blockIdx.z, h = blockIdx.y;
  const int kc = blockIdx.x * 128 + wave * 16;
  const int lm  = lane & 15;
  const int akb = (lane < 16) ? 0 : 8;
  const int bkb = (lane < 16) ? 0 : 16;
  const float scale = 0.125f; // 1/sqrt(64)

  // B = K^T for this wave's 16 columns: B(dk, kcol) = K[kcol][dk];
  // per lane kcol = kc+lm fixed, K pairs along dk -> contiguous 32B.
  const bf16* krow = kb + (size_t)(b * Sm + kc + lm) * Dm + h * DKm;
  const v16bf bk0 = *(const v16bf*)(krow + 0  + bkb);
  const v16bf bk1 = *(const v16bf*)(krow + 32 + bkb);

  float rm = -3.0e38f, rl = 0.0f;
  for (int qt = 0; qt < Sm; qt += 16) {
    const bf16* arow = qb + (size_t)(b * Sm + qt + lm) * Dm + h * DKm;
    v16bf a0 = load_afrag(arow, 0, akb);
    v16bf a1 = load_afrag(arow, 32, akb);
    v8f s = {};
    s = wmma_bf16(a0, bk0, s);
    s = wmma_bf16(a1, bk1, s);

    float tm = -3.0e38f;
#pragma unroll
    for (int r = 0; r < 8; ++r) tm = fmaxf(tm, s[r] * scale);
    const float nm = fmaxf(rm, tm);
    float add = 0.0f;
#pragma unroll
    for (int r = 0; r < 8; ++r) add += __expf(s[r] * scale - nm);
    rl = rl * __expf(rm - nm) + add;
    rm = nm;
  }

  const float om = __shfl_xor(rm, 16);
  const float ol = __shfl_xor(rl, 16);
  const float fm = fmaxf(rm, om);
  const float fl = rl * __expf(rm - fm) + ol * __expf(om - fm);
  if (lane < 16) {
    const size_t idx = (size_t)(b * NHm + h) * Sm + kc + lm;
    mcol[idx] = fm;
    invl[idx] = 1.0f / fl;
  }
}

// ---------------------------------------------------------------------------
// 4) heads[b][h][q][dk] = sum_k p(q,k) * V[k][dk],  p = exp(s*scale - m_k)/l_k.
// Wave -> 16 q rows x 64 dk. k-loop step 32: recompute 16x32 score tile
// (4 WMMA), normalize per key column (per-lane scalar m/1/l loads), repack
// f32 C-layout -> bf16 A-layout through per-wave LDS (ds_store_b16 ->
// ds_load_b128), then 4 WMMA against Vt B-fragments.
// ---------------------------------------------------------------------------
__global__ __launch_bounds__(256) void attn_heads(const bf16* __restrict__ qb,
                                                  const bf16* __restrict__ kb,
                                                  const bf16* __restrict__ vtb,
                                                  const float* __restrict__ mcol,
                                                  const float* __restrict__ invl,
                                                  bf16* __restrict__ headsb) {
  const int lane = threadIdx.x & 31;
  const int wave = threadIdx.x >> 5;
  const int b = blockIdx.z, h = blockIdx.y;
  const int q0 = blockIdx.x * 128 + wave * 16;
  const int lm  = lane & 15;
  const int akb = (lane < 16) ? 0 : 8;
  const int bkb = (lane < 16) ? 0 : 16;
  const int rowoff = (lane < 16) ? 0 : 8;
  const float scale = 0.125f;
  const size_t hsb = (size_t)(b * NHm + h) * Sm;

  // Per-wave P staging tile: 16 q-rows x 32 k-cols, padded stride 40 (80B,
  // multiple of 16B so ds_load_b128 stays aligned).
  __shared__ __align__(16) bf16 plds[8][16][40];

  // Q fragments for this wave's 16 rows (resident across the whole k loop).
  const bf16* qrow = qb + (size_t)(b * Sm + q0 + lm) * Dm + h * DKm;
  const v16bf qa0 = load_afrag(qrow, 0, akb);
  const v16bf qa1 = load_afrag(qrow, 32, akb);

  v8f hacc[4] = {};

  for (int kc = 0; kc < Sm; kc += 32) {
    // ---- scores: two 16x16 tiles over this 32-wide k chunk
    v8f sacc[2] = {};
#pragma unroll
    for (int j = 0; j < 2; ++j) {
      const int kcol = kc + j * 16 + lm;
      const bf16* krow = kb + (size_t)(b * Sm + kcol) * Dm + h * DKm;
      v16bf bk0 = *(const v16bf*)(krow + 0  + bkb);
      v16bf bk1 = *(const v16bf*)(krow + 32 + bkb);
      sacc[j] = wmma_bf16(qa0, bk0, sacc[j]);
      sacc[j] = wmma_bf16(qa1, bk1, sacc[j]);
    }

    // ---- normalize per key column, write P to LDS in row-major q x k
#pragma unroll
    for (int j = 0; j < 2; ++j) {
      const int kcol = kc + j * 16 + lm;
      const float mc = mcol[hsb + kcol];
      const float il = invl[hsb + kcol];
#pragma unroll
      for (int r = 0; r < 8; ++r) {
        const float p = __expf(sacc[j][r] * scale - mc) * il;
        plds[wave][rowoff + r][j * 16 + lm] = f2bf(p);
      }
    }
    // LDS is per-wave private; DS ops from one wave execute in order, so the
    // RAW store->load below is safe without a workgroup barrier.

    // ---- reload P as an A fragment (16x32)
    U256 pu;
    pu.lo = *(const U128*)&plds[wave][lm][2 * akb];
    pu.hi = *(const U128*)&plds[wave][lm][2 * akb + 8];
    const v16bf pa = __builtin_bit_cast(v16bf, pu);

    // ---- P @ V: Vt[b][h][dk][s] makes B(k,dk) pairs contiguous along s
#pragma unroll
    for (int t = 0; t < 4; ++t) {
      const int dk = t * 16 + lm;
      const bf16* vrow = vtb + ((size_t)(b * NHm + h) * DKm + dk) * Sm;
      v16bf bv = *(const v16bf*)(vrow + kc + bkb);
      hacc[t] = wmma_bf16(pa, bv, hacc[t]);
    }
  }

  // ---- store heads, flat [b][h][q][dk] (== reference reshape(b, s, d))
#pragma unroll
  for (int t = 0; t < 4; ++t) {
#pragma unroll
    for (int r = 0; r < 8; ++r) {
      const int q  = q0 + rowoff + r;
      const int dk = t * 16 + lm;
      headsb[(hsb + q) * DKm + dk] = f2bf(hacc[t][r]);
    }
  }
}

// ---------------------------------------------------------------------------
// launch
// ---------------------------------------------------------------------------
extern "C" void kernel_launch(void* const* d_in, const int* in_sizes, int n_in,
                              void* d_out, int out_size, void* d_ws, size_t ws_size,
                              hipStream_t stream) {
  (void)in_sizes; (void)n_in; (void)out_size; (void)ws_size;

  const float* x  = (const float*)d_in[0];
  const float* Wq = (const float*)d_in[1];
  const float* Wk = (const float*)d_in[2];
  const float* Wv = (const float*)d_in[3];
  const float* W0 = (const float*)d_in[4];
  float* out = (float*)d_out;

  const size_t XN = (size_t)Mtot * Dm;      // 6,291,456 elems
  const size_t WN = (size_t)Dm * Dm;        //   589,824 elems
  const size_t SN = (size_t)Bm * NHm * Sm;  //    98,304 elems

  char* p = (char*)d_ws;
  auto alloc = [&](size_t bytes) -> void* {
    void* r = (void*)p;
    p += (bytes + 255) & ~(size_t)255;
    return r;
  };
  bf16*  xb     = (bf16*)alloc(XN * sizeof(bf16));
  bf16*  wqb    = (bf16*)alloc(WN * sizeof(bf16));
  bf16*  wkb    = (bf16*)alloc(WN * sizeof(bf16));
  bf16*  wvb    = (bf16*)alloc(WN * sizeof(bf16));
  bf16*  w0b    = (bf16*)alloc(WN * sizeof(bf16));
  bf16*  qbuf   = (bf16*)alloc(XN * sizeof(bf16));
  bf16*  kbuf   = (bf16*)alloc(XN * sizeof(bf16));
  bf16*  vtb    = (bf16*)alloc(XN * sizeof(bf16));
  bf16*  headsb = (bf16*)alloc(XN * sizeof(bf16));
  float* mcol   = (float*)alloc(SN * sizeof(float));
  float* invl   = (float*)alloc(SN * sizeof(float));

  auto cvt = [&](const float* src, bf16* dst, size_t n) {
    int ni = (int)n;
    cvt_f32_bf16<<<dim3((ni + 255) / 256), dim3(256), 0, stream>>>(src, dst, ni);
  };
  cvt(x,  xb,  XN);
  cvt(Wq, wqb, WN);
  cvt(Wk, wkb, WN);
  cvt(Wv, wvb, WN);
  cvt(W0, w0b, WN);

  const dim3 gGemm(Mtot / 128, Dm / 64);   // (64, 12)
  const dim3 gAttn(Sm / 128, NHm, Bm);     // (16, 12, 4)
  const dim3 blk(256);

  gemm_xwt<1><<<gGemm, blk, 0, stream>>>(xb, wqb, nullptr, qbuf);
  gemm_xwt<1><<<gGemm, blk, 0, stream>>>(xb, wkb, nullptr, kbuf);
  gemm_xwt<2><<<gGemm, blk, 0, stream>>>(xb, wvb, nullptr, vtb);

  col_softmax_stats<<<gAttn, blk, 0, stream>>>(qbuf, kbuf, mcol, invl);
  attn_heads<<<gAttn, blk, 0, stream>>>(qbuf, kbuf, vtb, mcol, invl, headsb);

  gemm_xwt<0><<<gGemm, blk, 0, stream>>>(headsb, w0b, out, nullptr);
}
